// actionDecode_with_relu_32779190403209
// MI455X (gfx1250) — matrix operand order repaired and checked by reading, not compiled, not run on hardware
//
#include <hip/hip_runtime.h>

// Fused grouped MLP: out[b,n,:] = relu(x[b,n,:] @ W1[n] + b1[n]) @ W2[n] + b2[n]
// B=4096, N=8, D=1024, H=2048, A=256.
// ~172 GFLOP vs ~250MB HBM traffic (687 FLOP/B) -> compute bound; bf16 WMMA path.
// Software-pipelined: double-buffered LDS tiles, 1 barrier per k-step, global
// prefetch of the next tile overlaps the current WMMA block.

#define B_TOT  4096
#define NGRP   8
#define DMODEL 1024
#define HIDDEN 2048
#define ABIN   256

#define BM     64      // batch rows per workgroup
#define CHUNK  128     // hidden chunk kept in LDS
#define KSTEP  32      // WMMA K per step (bf16)
#define NK1    (DMODEL / KSTEP)   // 32
#define NK2    (CHUNK  / KSTEP)   // 4

// LDS strides (elements). 40 elems = 80B rows: 16B-aligned, 20-bank step -> conflict free.
#define XT_S   40
#define W1T_S  40
#define W2T_S  40
#define H_S    136     // 128 + 8 pad; 272B rows, 16B aligned

typedef __attribute__((ext_vector_type(16))) __bf16 v16bf;
typedef __attribute__((ext_vector_type(8)))  __bf16 v8bf;
typedef __attribute__((ext_vector_type(4)))  __bf16 v4bf;
typedef __attribute__((ext_vector_type(8)))  float  v8f;

static __device__ inline v16bf cat8(v8bf lo, v8bf hi) {
  return __builtin_shufflevector(lo, hi, 0,1,2,3,4,5,6,7,8,9,10,11,12,13,14,15);
}

__global__ __launch_bounds__(256)
void fused_group_mlp_wmma(const float* __restrict__ x,
                          const float* __restrict__ W1,
                          const float* __restrict__ b1,
                          const float* __restrict__ W2,
                          const float* __restrict__ b2,
                          float* __restrict__ out)
{
  // LDS: 2*5120 + 2*10240 + 2*20480 + 17408 = 89,088 B
  __shared__ __bf16 xt [2][BM    * XT_S ];  // x tile   [64 m][32 k]   (A layout)
  __shared__ __bf16 w1t[2][CHUNK * W1T_S];  // W1 tile  [128 n][32 k]  (B layout, transposed)
  __shared__ __bf16 w2t[2][ABIN  * W2T_S];  // W2 tile  [256 a][32 k]  (B layout, transposed)
  __shared__ __bf16 hs [BM * H_S];          // relu(h)  [64 m][128 n]  (A operand of GEMM2)

  const int t    = threadIdx.x;             // 256 threads = 8 wave32
  const int mb   = blockIdx.x;              // 64 M-blocks
  const int g    = blockIdx.y;              // 8 groups
  const int lane = t & 31;
  const int wv   = t >> 5;
  const int half = lane >> 4;               // K half for A/B fragments
  const int l16  = lane & 15;

  // GEMM1: wave -> 16 M rows x 64 hidden cols; GEMM2: wave -> 16 M rows x 128 bins
  const int mt  = wv & 3;                   // M tile (16 rows)
  const int nh1 = wv >> 2;                  // hidden half
  const int ah2 = wv >> 2;                  // action half

  const float* xg  = x  + ((size_t)(mb * BM) * NGRP + g) * DMODEL;
  const float* w1g = W1 + (size_t)g * DMODEL * HIDDEN;
  const float* w2g = W2 + (size_t)g * HIDDEN * ABIN;

  // Per-thread staging coordinates (fixed across iterations)
  const int xm  = t >> 3,          xk4 = t & 7;    // x: 2 float4/thread
  const int w1k = t >> 5,          w1n = t & 31;   // W1: 4 float4/thread (k += 8/iter)
  const int w2k = t >> 6,          w2a = t & 63;   // W2: 8 float4/thread (k += 4/iter)

  const v8f vzero = {0.f,0.f,0.f,0.f,0.f,0.f,0.f,0.f};

  v8f c2[8];                                // output accum, lives across chunks
  #pragma unroll
  for (int i = 0; i < 8; ++i) c2[i] = vzero;

  for (int ch = 0; ch < HIDDEN / CHUNK; ++ch) {
    // ================= GEMM1: h_chunk = x(64x1024) @ W1[:, ch*128 .. +128) =================
    v8f c1[4];
    #pragma unroll
    for (int i = 0; i < 4; ++i) c1[i] = vzero;

    float4 xr[2], wr[4];
    // prologue prefetch: k-tile 0
    #pragma unroll
    for (int i = 0; i < 2; ++i)
      xr[i] = *(const float4*)(xg + (size_t)(xm + i * 32) * (NGRP * DMODEL) + xk4 * 4);
    #pragma unroll
    for (int i = 0; i < 4; ++i)
      wr[i] = *(const float4*)(w1g + (size_t)(w1k + i * 8) * HIDDEN + ch * CHUNK + w1n * 4);

    for (int kbi = 0; kbi < NK1; ++kbi) {
      __bf16* xtp  = xt [kbi & 1];
      __bf16* w1tp = w1t[kbi & 1];

      // store prefetched tile -> LDS (fp32 -> bf16)
      #pragma unroll
      for (int i = 0; i < 2; ++i) {
        v4bf p; p[0] = (__bf16)xr[i].x; p[1] = (__bf16)xr[i].y;
                p[2] = (__bf16)xr[i].z; p[3] = (__bf16)xr[i].w;
        *(v4bf*)(xtp + (xm + i * 32) * XT_S + xk4 * 4) = p;
      }
      #pragma unroll
      for (int i = 0; i < 4; ++i) {
        const int k = w1k + i * 8;
        w1tp[(w1n * 4 + 0) * W1T_S + k] = (__bf16)wr[i].x;
        w1tp[(w1n * 4 + 1) * W1T_S + k] = (__bf16)wr[i].y;
        w1tp[(w1n * 4 + 2) * W1T_S + k] = (__bf16)wr[i].z;
        w1tp[(w1n * 4 + 3) * W1T_S + k] = (__bf16)wr[i].w;
      }
      __syncthreads();   // tile kbi visible; readers of the other buffer (kbi-1) are done

      // prefetch k-tile kbi+1 (overlaps the WMMA block below)
      if (kbi + 1 < NK1) {
        const int kb = (kbi + 1) * KSTEP;
        #pragma unroll
        for (int i = 0; i < 2; ++i)
          xr[i] = *(const float4*)(xg + (size_t)(xm + i * 32) * (NGRP * DMODEL) + kb + xk4 * 4);
        #pragma unroll
        for (int i = 0; i < 4; ++i)
          wr[i] = *(const float4*)(w1g + (size_t)(kb + w1k + i * 8) * HIDDEN
                                       + ch * CHUNK + w1n * 4);
      }

      // A fragment: 16x32 bf16 (lane half holds K {h*8..+7, 16+h*8..+7})
      const int am = mt * 16 + l16;
      const v8bf alo = *(const v8bf*)(xtp + am * XT_S + half * 8);
      const v8bf ahi = *(const v8bf*)(xtp + am * XT_S + 16 + half * 8);
      const v16bf a  = cat8(alo, ahi);

      #pragma unroll
      for (int nt = 0; nt < 4; ++nt) {
        const int ncol = nh1 * 64 + nt * 16 + l16;       // B: lane = column, K {h*8..+7, 16+h*8..+7}
        const v8bf blo = *(const v8bf*)(w1tp + ncol * W1T_S + half * 8);
        const v8bf bhi = *(const v8bf*)(w1tp + ncol * W1T_S + 16 + half * 8);
        c1[nt] = __builtin_amdgcn_wmma_f32_16x16x32_bf16(false, a, false, cat8(blo, bhi),
                                                         (short)0, c1[nt], false, false);
      }
    }

    // prologue prefetch for GEMM2 k-tile 0 (overlaps the h epilogue below)
    float4 w2r[8];
    #pragma unroll
    for (int i = 0; i < 8; ++i)
      w2r[i] = *(const float4*)(w2g + (size_t)(ch * CHUNK + w2k + i * 4) * ABIN + w2a * 4);

    // bias + ReLU -> bf16 h in LDS (C layout: lane=N, vgpr r -> M = r + 8*half).
    // hs readers (previous chunk's GEMM2) are long since past GEMM1's barriers.
    #pragma unroll
    for (int nt = 0; nt < 4; ++nt) {
      const int ncol  = nh1 * 64 + nt * 16 + l16;
      const float bia = b1[g * HIDDEN + ch * CHUNK + ncol];
      #pragma unroll
      for (int r = 0; r < 8; ++r) {
        const float v = fmaxf(c1[nt][r] + bia, 0.f);
        hs[(mt * 16 + half * 8 + r) * H_S + ncol] = (__bf16)v;
      }
    }

    // ================= GEMM2: c2 += h_chunk(64x128) @ W2[ch*128.., :256) =================
    for (int kbi = 0; kbi < NK2; ++kbi) {
      __bf16* w2tp = w2t[kbi & 1];

      #pragma unroll
      for (int i = 0; i < 8; ++i) {
        const int k = w2k + i * 4;
        w2tp[(w2a * 4 + 0) * W2T_S + k] = (__bf16)w2r[i].x;
        w2tp[(w2a * 4 + 1) * W2T_S + k] = (__bf16)w2r[i].y;
        w2tp[(w2a * 4 + 2) * W2T_S + k] = (__bf16)w2r[i].z;
        w2tp[(w2a * 4 + 3) * W2T_S + k] = (__bf16)w2r[i].w;
      }
      __syncthreads();   // also makes this chunk's hs visible before first use

      if (kbi + 1 < NK2) {
        const int kb2 = (kbi + 1) * KSTEP;
        #pragma unroll
        for (int i = 0; i < 8; ++i)
          w2r[i] = *(const float4*)(w2g + (size_t)(ch * CHUNK + kb2 + w2k + i * 4) * ABIN
                                        + w2a * 4);
      }

      const int kb2 = kbi * KSTEP;
      const int am  = mt * 16 + l16;
      const v8bf alo = *(const v8bf*)(hs + am * H_S + kb2 + half * 8);
      const v8bf ahi = *(const v8bf*)(hs + am * H_S + kb2 + 16 + half * 8);
      const v16bf a2 = cat8(alo, ahi);

      #pragma unroll
      for (int at = 0; at < 8; ++at) {
        const int acol = ah2 * 128 + at * 16 + l16;
        const v8bf blo = *(const v8bf*)(w2tp + acol * W2T_S + half * 8);
        const v8bf bhi = *(const v8bf*)(w2tp + acol * W2T_S + 16 + half * 8);
        c2[at] = __builtin_amdgcn_wmma_f32_16x16x32_bf16(false, a2, false, cat8(blo, bhi),
                                                         (short)0, c2[at], false, false);
      }
    }
    __syncthreads();     // chunk boundary: next chunk overwrites hs / reuses buffers
  }

  // ================= epilogue: + b2, store fp32 =================
  #pragma unroll
  for (int at = 0; at < 8; ++at) {
    const int acol  = ah2 * 128 + at * 16 + l16;
    const float bia = b2[g * ABIN + acol];
    #pragma unroll
    for (int r = 0; r < 8; ++r) {
      const int m = mt * 16 + half * 8 + r;
      out[((size_t)(mb * BM + m) * NGRP + g) * ABIN + acol] = c2[at][r] + bia;
    }
  }
}

extern "C" void kernel_launch(void* const* d_in, const int* in_sizes, int n_in,
                              void* d_out, int out_size, void* d_ws, size_t ws_size,
                              hipStream_t stream) {
  const float* x  = (const float*)d_in[0];
  const float* W1 = (const float*)d_in[1];
  const float* b1 = (const float*)d_in[2];
  const float* W2 = (const float*)d_in[3];
  const float* b2 = (const float*)d_in[4];
  float* out = (float*)d_out;
  (void)in_sizes; (void)n_in; (void)out_size; (void)d_ws; (void)ws_size;

  dim3 grid(B_TOT / BM, NGRP);   // 64 x 8 = 512 workgroups
  dim3 block(256);               // 8 wave32
  fused_group_mlp_wmma<<<grid, block, 0, stream>>>(x, W1, b1, W2, b2, out);
}
